// CausalSelfAttention_10900626997561
// MI455X (gfx1250) — matrix operand order repaired and checked by reading, not compiled, not run on hardware
//
#include <hip/hip_runtime.h>

// Problem constants (match reference)
#define Bv 4
#define Sv 2048
#define Dv 2048
#define Hv 16
#define HDv 128
#define N3D 6144   // 3*D

typedef __attribute__((ext_vector_type(16))) __bf16 v16bf;
typedef __attribute__((ext_vector_type(8)))  __bf16 v8bf;
typedef __attribute__((ext_vector_type(8)))  float  v8f;
typedef int v4i_t __attribute__((vector_size(16)));

#define AS1 __attribute__((address_space(1)))
#define AS3 __attribute__((address_space(3)))

union V16U { v16bf v; v8bf h[2]; };

__device__ __forceinline__ v16bf load16(const __bf16* p0, const __bf16* p1) {
  V16U u;
  u.h[0] = *(const v8bf*)p0;
  u.h[1] = *(const v8bf*)p1;
  return u.v;
}

__device__ __forceinline__ v8f vzero8() {
  v8f z = {0.f,0.f,0.f,0.f,0.f,0.f,0.f,0.f};
  return z;
}

__device__ __forceinline__ v8f wmma_bf16(v16bf a, v16bf b, v8f c) {
  // D = A(16x32 bf16) * B(32x16 bf16) + C(16x16 f32)
  return __builtin_amdgcn_wmma_f32_16x16x32_bf16(false, a, false, b, (short)0, c, false, false);
}

// --- async global->LDS staging (CDNA5 GLOBAL_LOAD_ASYNC_TO_LDS_B128) with
// --- guaranteed-compile fallback to load+ds_store if the builtin is absent.
#if __has_builtin(__builtin_amdgcn_global_load_async_to_lds_b128)
#define HAVE_ASYNC_LDS 1
__device__ __forceinline__ void cp16_async(const __bf16* g, __bf16* l) {
  AS1 char* gp = (AS1 char*)(__bf16*)g;   // generic -> AS1 (global)
  AS3 char* lp = (AS3 char*)l;            // generic -> AS3 (LDS)
  __builtin_amdgcn_global_load_async_to_lds_b128(
      (AS1 v4i_t*)gp, (AS3 v4i_t*)lp, 0, 0);
}
__device__ __forceinline__ void wait_async() {
#if __has_builtin(__builtin_amdgcn_s_wait_asynccnt)
  __builtin_amdgcn_s_wait_asynccnt(0);
#else
  asm volatile("s_wait_asynccnt 0" ::: "memory");
#endif
}
#else
#define HAVE_ASYNC_LDS 0
__device__ __forceinline__ void cp16_async(const __bf16* g, __bf16* l) {
  *(v8bf*)l = *(const v8bf*)g;
}
__device__ __forceinline__ void wait_async() {}
#endif

// ---------------------------------------------------------------------------
// Kernel 1: fp32 -> bf16 elementwise convert
// ---------------------------------------------------------------------------
__global__ void k_cvt_bf16(const float* __restrict__ in, __bf16* __restrict__ out, int n) {
  int i = blockIdx.x * blockDim.x + threadIdx.x;
  if (i < n) out[i] = (__bf16)in[i];
}

// ---------------------------------------------------------------------------
// Kernel 2: fp32 (R x C) -> bf16 transposed (C x R), tiled through LDS
// ---------------------------------------------------------------------------
__global__ __launch_bounds__(256) void k_transpose_cvt(const float* __restrict__ in,
                                                       __bf16* __restrict__ out,
                                                       int R, int C) {
  __shared__ float tile[32][33];
  int c0 = blockIdx.x * 32, r0 = blockIdx.y * 32;
  int tx = threadIdx.x, ty = threadIdx.y;   // (32, 8)
#pragma unroll
  for (int i = 0; i < 4; i++)
    tile[ty + 8 * i][tx] = in[(size_t)(r0 + ty + 8 * i) * C + c0 + tx];
  __syncthreads();
#pragma unroll
  for (int i = 0; i < 4; i++)
    out[(size_t)(c0 + ty + 8 * i) * R + r0 + tx] = (__bf16)tile[tx][ty + 8 * i];
}

// ---------------------------------------------------------------------------
// Kernel 3: QKV GEMM  (M=B*S=8192, K=D, N=3D) with scatter epilogue.
//   A  = xb  (M x K, bf16 row-major)
//   Bt = wqkvT (N x K, bf16)  -> B[k][n] = Bt[n][k]
// Workgroup: 256 thr (8 waves), tile 64(M) x 128(N), K-step 32 via LDS.
// Each wave: 32x32 (2x2 WMMA tiles).
// ---------------------------------------------------------------------------
__global__ __launch_bounds__(256) void k_qkv_gemm(const __bf16* __restrict__ xb,
                                                  const __bf16* __restrict__ wT,
                                                  float* __restrict__ q_raw,
                                                  float* __restrict__ k_raw,
                                                  float* __restrict__ v_out,
                                                  __bf16* __restrict__ vTb) {
  const int K = Dv;
  __shared__ __align__(16) __bf16 As[64][40];
  __shared__ __align__(16) __bf16 Bs[128][40];

  int tileN = blockIdx.x * 128;
  int tileM = blockIdx.y * 64;
  int t = threadIdx.x;
  int w = t >> 5, lane = t & 31;
  int lm = lane & 15, hi = lane >> 4, hi8 = hi * 8;
  int wm = w & 1, wn = w >> 1;

  v8f acc[2][2];
#pragma unroll
  for (int a = 0; a < 2; a++)
#pragma unroll
    for (int b = 0; b < 2; b++) acc[a][b] = vzero8();

  int arow = t >> 2, acol = (t & 3) * 8;
  int brow = t >> 1, bcol = (t & 1) * 16;
  const __bf16* Aptr = xb + (size_t)(tileM + arow) * K + acol;
  const __bf16* Bptr = wT + (size_t)(tileN + brow) * K + bcol;

  for (int kb = 0; kb < K; kb += 32) {
    cp16_async(Aptr,     &As[arow][acol]);
    cp16_async(Bptr,     &Bs[brow][bcol]);
    cp16_async(Bptr + 8, &Bs[brow][bcol + 8]);
    Aptr += 32; Bptr += 32;
    __builtin_prefetch(Aptr + 32, 0, 3);
    __builtin_prefetch(Bptr + 32, 0, 3);
    wait_async();
    __syncthreads();

    v16bf a0 = load16(&As[wm * 32 + lm][hi * 16],      &As[wm * 32 + lm][hi * 16 + 8]);
    v16bf a1 = load16(&As[wm * 32 + 16 + lm][hi * 16], &As[wm * 32 + 16 + lm][hi * 16 + 8]);
    v16bf b0 = load16(&Bs[wn * 32 + lm][hi * 16],      &Bs[wn * 32 + lm][hi * 16 + 8]);
    v16bf b1 = load16(&Bs[wn * 32 + 16 + lm][hi * 16], &Bs[wn * 32 + 16 + lm][hi * 16 + 8]);

    acc[0][0] = wmma_bf16(a0, b0, acc[0][0]);
    acc[0][1] = wmma_bf16(a0, b1, acc[0][1]);
    acc[1][0] = wmma_bf16(a1, b0, acc[1][0]);
    acc[1][1] = wmma_bf16(a1, b1, acc[1][1]);
    __syncthreads();
  }

  // Epilogue: scatter q/k raw to ws, v to d_out (+ bf16 V^T for attention).
#pragma unroll
  for (int mi = 0; mi < 2; mi++) {
#pragma unroll
    for (int ni = 0; ni < 2; ni++) {
      v8f c = acc[mi][ni];
#pragma unroll
      for (int j = 0; j < 8; j++) {
        int m = tileM + wm * 32 + mi * 16 + hi8 + j;
        int n = tileN + wn * 32 + ni * 16 + lm;
        float val = c[j];
        int b = m >> 11, s = m & (Sv - 1);
        int sec = n >> 11, dl = n & (Dv - 1);
        int h = dl >> 7, hd = dl & (HDv - 1);
        int bh = (b << 4) + h;
        size_t idx = ((size_t)bh * Sv + s) * HDv + hd;
        if (sec == 0) {
          q_raw[idx] = val;
        } else if (sec == 1) {
          k_raw[idx] = val;
        } else {
          v_out[idx] = val;
          vTb[((size_t)bh * HDv + hd) * Sv + s] = (__bf16)val;
        }
      }
    }
  }
}

// ---------------------------------------------------------------------------
// Kernel 4: RoPE on q and k. Emits bf16 q/k (B,H,S,HD) for attention and
// the fp32 rotated k into d_out.
// ---------------------------------------------------------------------------
__global__ void k_rope(const float* __restrict__ q_raw, const float* __restrict__ k_raw,
                       const float* __restrict__ cosT, const float* __restrict__ sinT,
                       __bf16* __restrict__ qb, __bf16* __restrict__ kb,
                       float* __restrict__ k_out) {
  int i = blockIdx.x * blockDim.x + threadIdx.x;
  if (i >= Bv * Hv * Sv * 64) return;
  int hd = i & 63;
  int s  = (i >> 6) & (Sv - 1);
  int bh = i >> 17;
  size_t base = ((size_t)bh * Sv + s) * HDv;
  float c1 = cosT[s * HDv + hd],      s1 = sinT[s * HDv + hd];
  float c2 = cosT[s * HDv + hd + 64], s2 = sinT[s * HDv + hd + 64];

  float q1 = q_raw[base + hd], q2 = q_raw[base + hd + 64];
  qb[base + hd]      = (__bf16)(q1 * c1 - q2 * s1);
  qb[base + hd + 64] = (__bf16)(q2 * c2 + q1 * s2);

  float k1 = k_raw[base + hd], k2 = k_raw[base + hd + 64];
  float ko1 = k1 * c1 - k2 * s1;
  float ko2 = k2 * c2 + k1 * s2;
  kb[base + hd]       = (__bf16)ko1;
  kb[base + hd + 64]  = (__bf16)ko2;
  k_out[base + hd]      = ko1;
  k_out[base + hd + 64] = ko2;
}

// ---------------------------------------------------------------------------
// Kernel 5: causal flash attention. One wave per (bh, 16 q-rows).
//   S = Q(16xHD) K^T, 32 keys/step, online softmax, ctx += P V.
// ---------------------------------------------------------------------------
__global__ __launch_bounds__(32) void k_attn(const __bf16* __restrict__ qb,
                                             const __bf16* __restrict__ kb,
                                             const __bf16* __restrict__ vTb,
                                             __bf16* __restrict__ ctxb) {
  __shared__ __align__(16) __bf16 Plds[16][40];
  int qt = blockIdx.x * 16;
  int bh = blockIdx.y;
  int lane = threadIdx.x;
  int lm = lane & 15, hi = lane >> 4, hi8 = hi * 8;

  const __bf16* qbase = qb  + ((size_t)bh * Sv + qt) * HDv;
  const __bf16* kbase = kb  + (size_t)bh * Sv * HDv;
  const __bf16* vbase = vTb + (size_t)bh * HDv * Sv;

  // Q A-fragments: lane = row lm, K chunks {hi*16.., hi*16+8..} per 32-wide chunk
  v16bf qf[4];
#pragma unroll
  for (int c = 0; c < 4; c++) {
    const __bf16* p = qbase + (size_t)lm * HDv + c * 32 + hi * 16;
    qf[c] = load16(p, p + 8);
  }

  v8f acc[8];
#pragma unroll
  for (int f = 0; f < 8; f++) acc[f] = vzero8();
  float mrun[8], lrun[8];
#pragma unroll
  for (int j = 0; j < 8; j++) { mrun[j] = -1e30f; lrun[j] = 0.f; }

  const float scale = 0.08838834764831845f;  // 1/sqrt(128)

  for (int kt = 0; kt <= qt + 15; kt += 32) {
    // ---- scores: two 16-key subtiles ----
    v8f s0 = vzero8(), s1 = vzero8();
#pragma unroll
    for (int c = 0; c < 4; c++) {
      const __bf16* p0 = kbase + (size_t)(kt + lm) * HDv + c * 32 + hi * 16;
      const __bf16* p1 = kbase + (size_t)(kt + 16 + lm) * HDv + c * 32 + hi * 16;
      s0 = wmma_bf16(qf[c], load16(p0, p0 + 8), s0);
      s1 = wmma_bf16(qf[c], load16(p1, p1 + 8), s1);
    }

    // ---- scale + causal mask + running max ----
    float mnew[8], al[8];
#pragma unroll
    for (int j = 0; j < 8; j++) {
      int m = qt + hi8 + j;
      float a = s0[j] * scale; if (kt + lm > m)      a = -1e30f;
      float b = s1[j] * scale; if (kt + 16 + lm > m) b = -1e30f;
      s0[j] = a; s1[j] = b;
      float r = fmaxf(a, b);
      r = fmaxf(r, __shfl_xor(r, 1));
      r = fmaxf(r, __shfl_xor(r, 2));
      r = fmaxf(r, __shfl_xor(r, 4));
      r = fmaxf(r, __shfl_xor(r, 8));
      mnew[j] = fmaxf(mrun[j], r);
      al[j] = __expf(mrun[j] - mnew[j]);
      mrun[j] = mnew[j];
    }

    // ---- P = exp(S - m), row sums, stash P to LDS ----
#pragma unroll
    for (int j = 0; j < 8; j++) {
      float p0 = __expf(s0[j] - mnew[j]);
      float p1 = __expf(s1[j] - mnew[j]);
      float sm = p0 + p1;
      sm += __shfl_xor(sm, 1);
      sm += __shfl_xor(sm, 2);
      sm += __shfl_xor(sm, 4);
      sm += __shfl_xor(sm, 8);
      lrun[j] = lrun[j] * al[j] + sm;
      Plds[hi8 + j][lm]      = (__bf16)p0;
      Plds[hi8 + j][16 + lm] = (__bf16)p1;
    }

    // ---- rescale context accumulators (row stats are lane-local) ----
#pragma unroll
    for (int f = 0; f < 8; f++)
#pragma unroll
      for (int j = 0; j < 8; j++) acc[f][j] *= al[j];

    __syncthreads();
    v16bf pf = load16(&Plds[lm][hi * 16], &Plds[lm][hi * 16 + 8]);

    // ---- ctx += P(16x32) * V(32xHD), HD in 8 column tiles ----
#pragma unroll
    for (int f = 0; f < 8; f++) {
      const __bf16* pv = vbase + (size_t)(f * 16 + lm) * Sv + kt + hi * 16;
      acc[f] = wmma_bf16(pf, load16(pv, pv + 8), acc[f]);
    }
    __syncthreads();
  }

  // ---- normalize and write ctx (B,S,D) as bf16 ----
  int b = bh >> 4, h = bh & 15;
  float inv[8];
#pragma unroll
  for (int j = 0; j < 8; j++) inv[j] = 1.0f / lrun[j];
#pragma unroll
  for (int f = 0; f < 8; f++) {
#pragma unroll
    for (int j = 0; j < 8; j++) {
      int m = qt + hi8 + j;
      ctxb[((size_t)(b * Sv + m)) * Dv + h * HDv + f * 16 + lm] = (__bf16)(acc[f][j] * inv[j]);
    }
  }
}

// ---------------------------------------------------------------------------
// Kernel 6: output GEMM  out = ctx (M x D) @ w_out (D x D), fp32 result.
// ---------------------------------------------------------------------------
__global__ __launch_bounds__(256) void k_out_gemm(const __bf16* __restrict__ ctxb,
                                                  const __bf16* __restrict__ wT,
                                                  float* __restrict__ out) {
  const int K = Dv;
  __shared__ __align__(16) __bf16 As[64][40];
  __shared__ __align__(16) __bf16 Bs[128][40];

  int tileN = blockIdx.x * 128;
  int tileM = blockIdx.y * 64;
  int t = threadIdx.x;
  int w = t >> 5, lane = t & 31;
  int lm = lane & 15, hi = lane >> 4, hi8 = hi * 8;
  int wm = w & 1, wn = w >> 1;

  v8f acc[2][2];
#pragma unroll
  for (int a = 0; a < 2; a++)
#pragma unroll
    for (int b = 0; b < 2; b++) acc[a][b] = vzero8();

  int arow = t >> 2, acol = (t & 3) * 8;
  int brow = t >> 1, bcol = (t & 1) * 16;
  const __bf16* Aptr = ctxb + (size_t)(tileM + arow) * K + acol;
  const __bf16* Bptr = wT + (size_t)(tileN + brow) * K + bcol;

  for (int kb = 0; kb < K; kb += 32) {
    cp16_async(Aptr,     &As[arow][acol]);
    cp16_async(Bptr,     &Bs[brow][bcol]);
    cp16_async(Bptr + 8, &Bs[brow][bcol + 8]);
    Aptr += 32; Bptr += 32;
    __builtin_prefetch(Aptr + 32, 0, 3);
    __builtin_prefetch(Bptr + 32, 0, 3);
    wait_async();
    __syncthreads();

    v16bf a0 = load16(&As[wm * 32 + lm][hi * 16],      &As[wm * 32 + lm][hi * 16 + 8]);
    v16bf a1 = load16(&As[wm * 32 + 16 + lm][hi * 16], &As[wm * 32 + 16 + lm][hi * 16 + 8]);
    v16bf b0 = load16(&Bs[wn * 32 + lm][hi * 16],      &Bs[wn * 32 + lm][hi * 16 + 8]);
    v16bf b1 = load16(&Bs[wn * 32 + 16 + lm][hi * 16], &Bs[wn * 32 + 16 + lm][hi * 16 + 8]);

    acc[0][0] = wmma_bf16(a0, b0, acc[0][0]);
    acc[0][1] = wmma_bf16(a0, b1, acc[0][1]);
    acc[1][0] = wmma_bf16(a1, b0, acc[1][0]);
    acc[1][1] = wmma_bf16(a1, b1, acc[1][1]);
    __syncthreads();
  }

#pragma unroll
  for (int mi = 0; mi < 2; mi++) {
#pragma unroll
    for (int ni = 0; ni < 2; ni++) {
      v8f c = acc[mi][ni];
#pragma unroll
      for (int j = 0; j < 8; j++) {
        int m = tileM + wm * 32 + mi * 16 + hi8 + j;
        int n = tileN + wn * 32 + ni * 16 + lm;
        out[(size_t)m * Dv + n] = c[j];
      }
    }
  }
}

// ---------------------------------------------------------------------------
// Host launcher
// ---------------------------------------------------------------------------
extern "C" void kernel_launch(void* const* d_in, const int* in_sizes, int n_in,
                              void* d_out, int out_size, void* d_ws, size_t ws_size,
                              hipStream_t stream) {
  (void)in_sizes; (void)n_in; (void)out_size; (void)ws_size;

  const float* x    = (const float*)d_in[0];
  const float* rc   = (const float*)d_in[1];
  const float* rs   = (const float*)d_in[2];
  const float* wqkv = (const float*)d_in[3];
  const float* wout = (const float*)d_in[4];

  float* out   = (float*)d_out;                          // (B,S,D)
  float* k_out = out + (size_t)Bv * Sv * Dv;             // (B,H,S,HD)
  float* v_out = k_out + (size_t)Bv * Hv * Sv * HDv;     // (B,H,S,HD)

  const size_t nBSD = (size_t)Bv * Sv * Dv;              // 16,777,216
  char* ws = (char*)d_ws;
  size_t off = 0;
  __bf16* xb    = (__bf16*)(ws + off); off += nBSD * 2;                 // x bf16
  __bf16* wqkvT = (__bf16*)(ws + off); off += (size_t)Dv * N3D * 2;     // (3D, D)
  __bf16* woutT = (__bf16*)(ws + off); off += (size_t)Dv * Dv * 2;      // (D, D)
  float*  q_raw = (float*)(ws + off);  off += nBSD * 4;                 // pre-rope q
  float*  k_raw = (float*)(ws + off);  off += nBSD * 4;                 // pre-rope k
  __bf16* qb    = (__bf16*)(ws + off); off += nBSD * 2;                 // rope'd q
  __bf16* kb    = (__bf16*)(ws + off); off += nBSD * 2;                 // rope'd k
  __bf16* vTb   = (__bf16*)(ws + off); off += nBSD * 2;                 // V^T (B,H,HD,S)
  __bf16* ctxb  = (__bf16*)(ws + off); off += nBSD * 2;                 // attention ctx

  // 1) converts / transposes
  k_cvt_bf16<<<(int)(nBSD / 256), 256, 0, stream>>>(x, xb, (int)nBSD);
  k_transpose_cvt<<<dim3(N3D / 32, Dv / 32), dim3(32, 8), 0, stream>>>(wqkv, wqkvT, Dv, N3D);
  k_transpose_cvt<<<dim3(Dv / 32, Dv / 32), dim3(32, 8), 0, stream>>>(wout, woutT, Dv, Dv);

  // 2) QKV projection
  k_qkv_gemm<<<dim3(N3D / 128, (Bv * Sv) / 64), 256, 0, stream>>>(
      xb, wqkvT, q_raw, k_raw, v_out, vTb);

  // 3) RoPE
  k_rope<<<(Bv * Hv * Sv * 64) / 256, 256, 0, stream>>>(
      q_raw, k_raw, rc, rs, qb, kb, k_out);

  // 4) causal flash attention
  k_attn<<<dim3(Sv / 16, Bv * Hv), 32, 0, stream>>>(qb, kb, vTb, ctxb);

  // 5) output projection
  k_out_gemm<<<dim3(Dv / 128, (Bv * Sv) / 64), 256, 0, stream>>>(ctxb, woutT, out);
}
